// rnn_phy_65455301591503
// MI455X (gfx1250) — matrix operand order repaired and checked by reading, not compiled, not run on hardware
//
#include <hip/hip_runtime.h>
#include <hip/hip_bf16.h>
#include <math.h>

typedef __attribute__((ext_vector_type(16))) _Float16 v16h;
typedef __attribute__((ext_vector_type(8)))  _Float16 v8h;
typedef __attribute__((ext_vector_type(8)))  float    v8f;
typedef __attribute__((ext_vector_type(4)))  float    v4f;

#define N_TRK 512
#define LHIST 20
#define PLEN  10
#define HDIM  128
#define EDIM  64
#define NHEAD 10
#define FFDIM 2048
#define DDIM  130   // H + 2
#define DPAD  160   // padded to multiple of 32 (K-chunk size)
#define DHEAD 13
#define DHP   32    // per-head K padded to one WMMA chunk
#define NLAY  3
#define DTs   0.1f

// ---------------------------------------------------------------------------
// Generic WMMA GEMM: C[z] = act(scale * A[z] x B[z]^T + bias [+ C]) [+ resid]
//   A: M x K (row stride lda, batch stride sAz);  B: [Nt x K] row-major.
//   REQUIREMENTS: K % 32 == 0; row strides / base offsets are multiples of
//   4 floats. Pad columns of at least one operand are zero.
//   Block = 4 waves = 64 output rows x 16*TN output cols. All 4 waves share
//   the SAME B columns, so the block cooperatively stages each 32-K B tile
//   in LDS (converted to f16 once), double-buffered with one barrier per
//   chunk. Waves read their fragments with ds_load_b128. A fragments are
//   per-wave (distinct rows) and stay as direct global float4 loads.
//   Row/col OOB lanes use clamped addresses (results never stored); EXEC
//   stays all-ones through every WMMA, as the ISA requires.
// ---------------------------------------------------------------------------
template<int TN>
__global__ void gemm_wmma_kernel(
    const float* A, int lda, long sAz,
    const float* B, int ldb, long sBz,
    float* __restrict__ C, int ldc, long sCz,
    const float* bias,
    const float* resid, int ldr,
    int M, int Nt, int K, float scale, int act, int acc)
{
  constexpr int COLS = 16 * TN;     // B columns staged per block
  constexpr int TPC  = 128 / COLS;  // fill threads per column
  constexpr int KPT  = 32 / TPC;    // k-values per fill thread
  constexpr int LSTR = 40;          // LDS column stride in halfs (bank skew)

  __shared__ _Float16 lsb[2][COLS][LSTR];

  const int lane = threadIdx.x & 31;
  const int wave = threadIdx.x >> 5;
  const int r    = lane & 15;
  const int hi   = lane >> 4;
  const int tm   = blockIdx.y * 64 + wave * 16;   // wave's output-row tile
  const int tnb  = blockIdx.x * COLS;             // block's output-col base
  A += (long)blockIdx.z * sAz;
  B += (long)blockIdx.z * sBz;
  C += (long)blockIdx.z * sCz;

  const int ra = min(tm + r, M - 1);              // clamped A row (OOB rows never stored)
  const float* Arow = A + (long)ra * lda;

  // Fill-role mapping: thread -> (column, k-range) of the staged B tile.
  const int fc = threadIdx.x % COLS;
  const int fk = (threadIdx.x / COLS) * KPT;
  const float* Frow = B + (long)min(tnb + fc, Nt - 1) * ldb;  // clamped B row

  auto fill = [&](int buf, int k0) {
#pragma unroll
    for (int q = 0; q < KPT / 4; ++q) {
      v4f f = *(const v4f*)(Frow + k0 + fk + q * 4);
      _Float16* d = &lsb[buf][fc][fk + q * 4];
      d[0] = (_Float16)f[0]; d[1] = (_Float16)f[1];
      d[2] = (_Float16)f[2]; d[3] = (_Float16)f[3];
    }
  };

  const v8f vzero = {0.f,0.f,0.f,0.f,0.f,0.f,0.f,0.f};
  v8f accv[TN];
#pragma unroll
  for (int t = 0; t < TN; ++t) accv[t] = vzero;

  const int nIter = K >> 5;
  fill(0, 0);
  __syncthreads();
  for (int i = 0; i < nIter; ++i) {
    const int k0 = i * 32;
    if (i + 1 < nIter) {
      fill((i + 1) & 1, k0 + 32);                 // stage next chunk
      __builtin_prefetch(&Arow[k0 + 32], 0, 1);   // uniform condition
    }
    // A fragment (global): e<8 -> k = hi*8+e ; e>=8 -> k = 16+hi*8+(e-8)
    v4f a0 = *(const v4f*)(Arow + k0 + hi * 8);
    v4f a1 = *(const v4f*)(Arow + k0 + hi * 8 + 4);
    v4f a2 = *(const v4f*)(Arow + k0 + 16 + hi * 8);
    v4f a3 = *(const v4f*)(Arow + k0 + 16 + hi * 8 + 4);
    v16h av;
#pragma unroll
    for (int e = 0; e < 4; ++e) {
      av[e]      = (_Float16)a0[e];
      av[e + 4]  = (_Float16)a1[e];
      av[e + 8]  = (_Float16)a2[e];
      av[e + 12] = (_Float16)a3[e];
    }
    // B fragments from LDS: lane holds col r of tile t; e<8 -> k = hi*8+e,
    // e>=8 -> k = 16+hi*8+(e-8) (same order as A) at [t*16+r][hi*8], [+16].
#pragma unroll
    for (int t = 0; t < TN; ++t) {
      const v8h* bp = (const v8h*)&lsb[i & 1][t * 16 + r][hi * 8];
      v8h b0 = bp[0], b1 = bp[2];
      v16h bv;
#pragma unroll
      for (int e = 0; e < 8; ++e) { bv[e] = b0[e]; bv[e + 8] = b1[e]; }
      accv[t] = __builtin_amdgcn_wmma_f32_16x16x32_f16(false, av, false, bv,
                                                       (short)0, accv[t], false, false);
    }
    __syncthreads();   // publish next buffer / protect refill of current
  }

  // Epilogue: C layout — lane holds col r, rows v + 8*hi. Guarded stores.
#pragma unroll
  for (int t = 0; t < TN; ++t) {
    int n = tnb + 16 * t + r;
#pragma unroll
    for (int v = 0; v < 8; ++v) {
      int m = tm + v + hi * 8;
      if (m < M && n < Nt) {
        float x = accv[t][v] * scale;
        if (bias)  x += bias[n];
        if (acc)   x += C[(long)m * ldc + n];
        if (act == 1) x = fmaxf(x, 0.f);
        if (resid) x += resid[(long)m * ldr + n];
        C[(long)m * ldc + n] = x;
      }
    }
  }
}

// Pack a [rows x K] matrix into [rows x Kp] with zero padding (once per call).
__global__ void pack_kpad_kernel(const float* __restrict__ src, float* __restrict__ dst,
                                 int rows, int K, int Kp)
{
  long i = (long)blockIdx.x * blockDim.x + threadIdx.x;
  if (i >= (long)rows * Kp) return;
  int r = (int)(i / Kp), k = (int)(i % Kp);
  dst[i] = (k < K) ? src[(long)r * K + k] : 0.f;
}

// Repack qkv (N x 390, ld 400) into per-head padded q/k [NHEAD][N][DHP]
// (cols 13..31 zero) and transposed v vT [NHEAD][DHEAD][N].
__global__ void repack_attn_kernel(const float* __restrict__ qkv,
                                   float* __restrict__ qp, float* __restrict__ kp,
                                   float* __restrict__ vT)
{
  int i = blockIdx.x * blockDim.x + threadIdx.x;   // NHEAD * N * DHP
  if (i >= NHEAD * N_TRK * DHP) return;
  int d = i & (DHP - 1);
  int n = (i >> 5) & (N_TRK - 1);
  int h = i / (DHP * N_TRK);
  const float* row = qkv + (long)n * 400;
  qp[i] = (d < DHEAD) ? row[h * DHEAD + d] : 0.f;
  kp[i] = (d < DHEAD) ? row[DDIM + h * DHEAD + d] : 0.f;
  if (d < DHEAD)
    vT[((long)h * DHEAD + d) * N_TRK + n] = row[2 * DDIM + h * DHEAD + d];
}

// ---------------------------------------------------------------------------
// Row softmax over length L (one 256-thread block per row)
// ---------------------------------------------------------------------------
__global__ void softmax_rows_kernel(float* __restrict__ x, int L)
{
  const int tid = threadIdx.x;
  float* row = x + (long)blockIdx.x * L;
  __shared__ float red[256];
  float m = -1e30f;
  for (int i = tid; i < L; i += 256) m = fmaxf(m, row[i]);
  red[tid] = m; __syncthreads();
  for (int s = 128; s > 0; s >>= 1) {
    if (tid < s) red[tid] = fmaxf(red[tid], red[tid + s]);
    __syncthreads();
  }
  m = red[0]; __syncthreads();
  float sum = 0.f;
  for (int i = tid; i < L; i += 256) { float e = expf(row[i] - m); row[i] = e; sum += e; }
  red[tid] = sum; __syncthreads();
  for (int s = 128; s > 0; s >>= 1) {
    if (tid < s) red[tid] += red[tid + s];
    __syncthreads();
  }
  float inv = 1.f / red[0];
  for (int i = tid; i < L; i += 256) row[i] *= inv;
}

// ---------------------------------------------------------------------------
// x = LayerNorm(x + r) * s + b over D=130 features (row stride DPAD),
// re-zeroing the pad columns [D, DPAD).  One block per row, 256 threads.
// ---------------------------------------------------------------------------
__global__ void add_ln_kernel(float* __restrict__ x, const float* __restrict__ rr,
                              const float* __restrict__ s, const float* __restrict__ b)
{
  const int tid = threadIdx.x;
  float* xr = x + (long)blockIdx.x * DPAD;
  const float* rp = rr + (long)blockIdx.x * DPAD;
  __shared__ float red[256];
  float t = 0.f;
  if (tid < DDIM) t = xr[tid] + rp[tid];
  red[tid] = t; __syncthreads();
  for (int st = 128; st > 0; st >>= 1) { if (tid < st) red[tid] += red[tid + st]; __syncthreads(); }
  float mean = red[0] / (float)DDIM; __syncthreads();
  float d = (tid < DDIM) ? (t - mean) : 0.f;
  red[tid] = d * d; __syncthreads();
  for (int st = 128; st > 0; st >>= 1) { if (tid < st) red[tid] += red[tid + st]; __syncthreads(); }
  float inv = rsqrtf(red[0] / (float)DDIM + 1e-5f);
  if (tid < DPAD) xr[tid] = (tid < DDIM) ? ((t - mean) * inv * s[tid] + b[tid]) : 0.f;
}

__device__ __forceinline__ float sigf(float x) { return 1.f / (1.f + expf(-x)); }

// LSTM gates: g = [i|f|gg|o] (N x 4H). Writes c (in place) and h2 into
// xbuf[:, 0:H] (row stride DPAD) — h2 is the first part of the encoder input.
__global__ void lstm_gate_kernel(const float* __restrict__ g, float* __restrict__ c,
                                 float* __restrict__ xbuf)
{
  int i = blockIdx.x * blockDim.x + threadIdx.x;
  if (i >= N_TRK * HDIM) return;
  int n = i / HDIM, j = i % HDIM;
  const float* gr = g + (long)n * (4 * HDIM);
  float c2 = sigf(gr[HDIM + j]) * c[i] + sigf(gr[j]) * tanhf(gr[2 * HDIM + j]);
  c[i] = c2;
  xbuf[(long)n * DPAD + j] = sigf(gr[3 * HDIM + j]) * tanhf(c2);
}

// xbuf[:, H:H+2] = inp[:, 0:2] (position), xbuf[:, D:DPAD] = 0 (padding)
__global__ void set_pos_pad_kernel(float* __restrict__ xbuf, const float* __restrict__ inp)
{
  int i = blockIdx.x * blockDim.x + threadIdx.x;   // N * 32 entries
  if (i >= N_TRK * 32) return;
  int n = i >> 5, col = HDIM + (i & 31);           // cols 128..159
  xbuf[(long)n * DPAD + col] = (col < DDIM) ? inp[n * 32 + (col - HDIM)] : 0.f;
}

// dst[n*wst + j] = (j<4) ? Tracks[n, t, 1+j] : 0      (wst = 32 for inp, 4 for cf)
__global__ void extract_inp_kernel(const float* __restrict__ tracks,
                                   float* __restrict__ dst, int t, int wst)
{
  int i = blockIdx.x * blockDim.x + threadIdx.x;
  if (i >= N_TRK * wst) return;
  int n = i / wst, j = i % wst;
  dst[i] = (j < 4) ? tracks[(long)n * LHIST * 5 + t * 5 + 1 + j] : 0.f;
}

__global__ void zero_kernel(float* __restrict__ p, int n)
{
  int i = blockIdx.x * blockDim.x + threadIdx.x;
  if (i < n) p[i] = 0.f;
}

// v = a*DT + cf[2:4]; p = a*DT^2/2 + v*DT + cf[0:2]; out=[p,v] ->
// writes next input (stride 32, pads stay zero), next cf, and d_out[n, t, :]
__global__ void pstep_integrate_kernel(const float* __restrict__ a, float* __restrict__ cf,
                                       float* __restrict__ inp, float* __restrict__ out, int t)
{
  int n = blockIdx.x * blockDim.x + threadIdx.x;
  if (n >= N_TRK) return;
  float a0 = a[n * 2 + 0], a1 = a[n * 2 + 1];
  float p0 = cf[n * 4 + 0], p1 = cf[n * 4 + 1];
  float v0 = cf[n * 4 + 2], v1 = cf[n * 4 + 3];
  float nv0 = a0 * DTs + v0,  nv1 = a1 * DTs + v1;
  float np0 = a0 * (DTs * DTs * 0.5f) + nv0 * DTs + p0;
  float np1 = a1 * (DTs * DTs * 0.5f) + nv1 * DTs + p1;
  cf[n * 4 + 0] = np0; cf[n * 4 + 1] = np1; cf[n * 4 + 2] = nv0; cf[n * 4 + 3] = nv1;
  inp[n * 32 + 0] = np0; inp[n * 32 + 1] = np1; inp[n * 32 + 2] = nv0; inp[n * 32 + 3] = nv1;
  long base = ((long)n * PLEN + t) * 4;
  out[base + 0] = np0; out[base + 1] = np1; out[base + 2] = nv0; out[base + 3] = nv1;
}

// ---------------------------------------------------------------------------
// Host-side orchestration
// ---------------------------------------------------------------------------
struct Weights {
  const float *Tracks, *W_emb, *b_emb, *W_ih, *W_hh, *b_ih, *b_hh;
  const float *qkv_w, *qkv_b, *out_w, *out_b, *ff1_w, *ff1_b, *ff2_w, *ff2_b;
  const float *ln1_s, *ln1_b, *ln2_s, *ln2_b;
  const float *W_sp, *b_sp, *W_p1, *b_p1, *W_p2, *b_p2, *W_pl, *b_pl;
};

struct WSBuf {
  float *Wq, *Wo, *Wf1, *Wsp, *Wemb;               // K-padded weight copies
  float *qkv, *qp, *kp, *vT, *scores, *ff;
  float *xbuf, *xres, *attout, *emb, *g;
  float *Sh, *c, *x1, *x2, *inp, *cf, *ab;
};

static void gemm(hipStream_t st, const float* A, int lda, long sAz,
                 const float* B, int ldb, long sBz,
                 float* C, int ldc, long sCz,
                 const float* bias, const float* resid, int ldr,
                 int M, int Nt, int K, float scale, int act, int acc, int Z)
{
  if (Nt >= 64) {
    dim3 grid((Nt + 63) / 64, (M + 63) / 64, Z);
    gemm_wmma_kernel<4><<<grid, dim3(128), 0, st>>>(A, lda, sAz, B, ldb, sBz,
                                                    C, ldc, sCz, bias, resid, ldr,
                                                    M, Nt, K, scale, act, acc);
  } else {
    dim3 grid((Nt + 15) / 16, (M + 63) / 64, Z);
    gemm_wmma_kernel<1><<<grid, dim3(128), 0, st>>>(A, lda, sAz, B, ldb, sBz,
                                                    C, ldc, sCz, bias, resid, ldr,
                                                    M, Nt, K, scale, act, acc);
  }
}

static void pack(hipStream_t st, const float* src, float* dst, int rows, int K, int Kp)
{
  long n = (long)rows * Kp;
  pack_kpad_kernel<<<dim3((unsigned)((n + 255) / 256)), dim3(256), 0, st>>>(src, dst, rows, K, Kp);
}

// 3-layer encoder over xbuf (512 x DPAD) -> Sh = x @ W_sp.T + b_sp
static void run_sp(hipStream_t st, const Weights& w, WSBuf& ws)
{
  const float inv_sqrt_dh = 0.277350098112615f;   // 1/sqrt(13)
  const int QLD = 400;                            // qkv buffer row stride (3D=390 padded)
  for (int j = 0; j < NLAY; ++j) {
    // qkv = x @ qkv_w.T + qkv_b   (K padded 130->160; weight pad is zero)
    gemm(st, ws.xbuf, DPAD, 0, ws.Wq + (long)j * 3 * DDIM * DPAD, DPAD, 0,
         ws.qkv, QLD, 0, w.qkv_b + (long)j * 3 * DDIM, nullptr, 0,
         N_TRK, 3 * DDIM, DPAD, 1.f, 0, 0, 1);
    // per-head padded q/k and transposed v
    repack_attn_kernel<<<dim3((NHEAD * N_TRK * DHP + 255) / 256), dim3(256), 0, st>>>(
        ws.qkv, ws.qp, ws.kp, ws.vT);
    // scores[h] = q_h @ k_h.T / sqrt(DH)   (K = 32, zero-padded both sides)
    gemm(st, ws.qp, DHP, (long)N_TRK * DHP, ws.kp, DHP, (long)N_TRK * DHP,
         ws.scores, N_TRK, (long)N_TRK * N_TRK, nullptr, nullptr, 0,
         N_TRK, N_TRK, DHP, inv_sqrt_dh, 0, 0, NHEAD);
    softmax_rows_kernel<<<dim3(NHEAD * N_TRK), dim3(256), 0, st>>>(ws.scores, N_TRK);
    // attout[:, h*13+d] = att_h @ v_h   (B = vT rows, contiguous K = 512)
    gemm(st, ws.scores, N_TRK, (long)N_TRK * N_TRK, ws.vT, N_TRK, (long)DHEAD * N_TRK,
         ws.attout, DPAD, DHEAD, nullptr, nullptr, 0,
         N_TRK, DHEAD, N_TRK, 1.f, 0, 0, NHEAD);
    // a = attout @ out_w.T + out_b   (attout pads zero; weight pad zero)
    gemm(st, ws.attout, DPAD, 0, ws.Wo + (long)j * DDIM * DPAD, DPAD, 0,
         ws.xres, DPAD, 0, w.out_b + (long)j * DDIM, nullptr, 0,
         N_TRK, DDIM, DPAD, 1.f, 0, 0, 1);
    add_ln_kernel<<<dim3(N_TRK), dim3(256), 0, st>>>(ws.xbuf, ws.xres,
        w.ln1_s + (long)j * DDIM, w.ln1_b + (long)j * DDIM);
    // ff = relu(x @ ff1_w.T + ff1_b)
    gemm(st, ws.xbuf, DPAD, 0, ws.Wf1 + (long)j * FFDIM * DPAD, DPAD, 0,
         ws.ff, FFDIM, 0, w.ff1_b + (long)j * FFDIM, nullptr, 0,
         N_TRK, FFDIM, DPAD, 1.f, 1, 0, 1);
    // f = ff @ ff2_w.T + ff2_b   (K = 2048 already a multiple of 32)
    gemm(st, ws.ff, FFDIM, 0, w.ff2_w + (long)j * DDIM * FFDIM, FFDIM, 0,
         ws.xres, DPAD, 0, w.ff2_b + (long)j * DDIM, nullptr, 0,
         N_TRK, DDIM, FFDIM, 1.f, 0, 0, 1);
    add_ln_kernel<<<dim3(N_TRK), dim3(256), 0, st>>>(ws.xbuf, ws.xres,
        w.ln2_s + (long)j * DDIM, w.ln2_b + (long)j * DDIM);
  }
  gemm(st, ws.xbuf, DPAD, 0, ws.Wsp, DPAD, 0, ws.Sh, HDIM, 0, w.b_sp, nullptr, 0,
       N_TRK, HDIM, DPAD, 1.f, 0, 0, 1);
}

// emb + LSTM (hidden input = hin) -> xbuf = [h2 | pos | 0-pad]
static void rm_lstm(hipStream_t st, const Weights& w, WSBuf& ws, const float* hin)
{
  gemm(st, ws.inp, 32, 0, ws.Wemb, 32, 0, ws.emb, EDIM, 0, w.b_emb, nullptr, 0,
       N_TRK, EDIM, 32, 1.f, 0, 0, 1);
  gemm(st, ws.emb, EDIM, 0, w.W_ih, EDIM, 0, ws.g, 4 * HDIM, 0, w.b_ih, nullptr, 0,
       N_TRK, 4 * HDIM, EDIM, 1.f, 0, 0, 1);
  gemm(st, hin, HDIM, 0, w.W_hh, HDIM, 0, ws.g, 4 * HDIM, 0, w.b_hh, nullptr, 0,
       N_TRK, 4 * HDIM, HDIM, 1.f, 0, 1, 1);
  lstm_gate_kernel<<<dim3((N_TRK * HDIM + 255) / 256), dim3(256), 0, st>>>(ws.g, ws.c, ws.xbuf);
  set_pos_pad_kernel<<<dim3((N_TRK * 32 + 255) / 256), dim3(256), 0, st>>>(ws.xbuf, ws.inp);
}

extern "C" void kernel_launch(void* const* d_in, const int* in_sizes, int n_in,
                              void* d_out, int out_size, void* d_ws, size_t ws_size,
                              hipStream_t stream)
{
  (void)in_sizes; (void)n_in; (void)out_size; (void)ws_size;
  Weights w;
  w.Tracks = (const float*)d_in[0];
  w.W_emb  = (const float*)d_in[1];   w.b_emb = (const float*)d_in[2];
  w.W_ih   = (const float*)d_in[3];   w.W_hh  = (const float*)d_in[4];
  w.b_ih   = (const float*)d_in[5];   w.b_hh  = (const float*)d_in[6];
  w.qkv_w  = (const float*)d_in[7];   w.qkv_b = (const float*)d_in[8];
  w.out_w  = (const float*)d_in[9];   w.out_b = (const float*)d_in[10];
  w.ff1_w  = (const float*)d_in[11];  w.ff1_b = (const float*)d_in[12];
  w.ff2_w  = (const float*)d_in[13];  w.ff2_b = (const float*)d_in[14];
  w.ln1_s  = (const float*)d_in[15];  w.ln1_b = (const float*)d_in[16];
  w.ln2_s  = (const float*)d_in[17];  w.ln2_b = (const float*)d_in[18];
  w.W_sp   = (const float*)d_in[19];  w.b_sp  = (const float*)d_in[20];
  w.W_p1   = (const float*)d_in[21];  w.b_p1  = (const float*)d_in[22];
  w.W_p2   = (const float*)d_in[23];  w.b_p2  = (const float*)d_in[24];
  w.W_pl   = (const float*)d_in[25];  w.b_pl  = (const float*)d_in[26];

  float* p = (float*)d_ws;
  WSBuf ws;
  ws.Wq     = p; p += (long)NLAY * 3 * DDIM * DPAD;
  ws.Wo     = p; p += (long)NLAY * DDIM * DPAD;
  ws.Wf1    = p; p += (long)NLAY * FFDIM * DPAD;
  ws.Wsp    = p; p += (long)HDIM * DPAD;
  ws.Wemb   = p; p += (long)EDIM * 32;
  ws.qkv    = p; p += (long)N_TRK * 400;
  ws.qp     = p; p += (long)NHEAD * N_TRK * DHP;
  ws.kp     = p; p += (long)NHEAD * N_TRK * DHP;
  ws.vT     = p; p += (long)NHEAD * DHEAD * N_TRK;
  ws.scores = p; p += (long)NHEAD * N_TRK * N_TRK;
  ws.ff     = p; p += (long)N_TRK * FFDIM;
  ws.xbuf   = p; p += (long)N_TRK * DPAD;
  ws.xres   = p; p += (long)N_TRK * DPAD;
  ws.attout = p; p += (long)N_TRK * DPAD;
  ws.emb    = p; p += (long)N_TRK * EDIM;
  ws.g      = p; p += (long)N_TRK * 4 * HDIM;
  ws.Sh     = p; p += (long)N_TRK * HDIM;
  ws.c      = p; p += (long)N_TRK * HDIM;
  ws.x1     = p; p += (long)N_TRK * HDIM;
  ws.x2     = p; p += (long)N_TRK * HDIM;
  ws.inp    = p; p += (long)N_TRK * 32;
  ws.cf     = p; p += (long)N_TRK * 4;
  ws.ab     = p; p += (long)N_TRK * 2;

  float* out = (float*)d_out;

  // ---- once per call: K-pad weight copies, clear state buffers ----
  pack(stream, w.qkv_w, ws.Wq,  NLAY * 3 * DDIM, DDIM, DPAD);
  pack(stream, w.out_w, ws.Wo,  NLAY * DDIM,     DDIM, DPAD);
  pack(stream, w.ff1_w, ws.Wf1, NLAY * FFDIM,    DDIM, DPAD);
  pack(stream, w.W_sp,  ws.Wsp, HDIM,            DDIM, DPAD);
  pack(stream, w.W_emb, ws.Wemb, EDIM, 4, 32);
  zero_kernel<<<dim3((N_TRK * HDIM + 255) / 256), dim3(256), 0, stream>>>(ws.c, N_TRK * HDIM);
  zero_kernel<<<dim3((N_TRK * DPAD + 255) / 256), dim3(256), 0, stream>>>(ws.attout, N_TRK * DPAD);

  // ---- initial step: h = c = 0, input = Tracks[:, 0, 1:] ----
  extract_inp_kernel<<<dim3((N_TRK * 32 + 255) / 256), dim3(256), 0, stream>>>(w.Tracks, ws.inp, 0, 32);
  rm_lstm(stream, w, ws, ws.c);   // c is still zero when read as hidden input
  run_sp(stream, w, ws);

  // ---- scan over the 20 history steps ----
  for (int t = 0; t < LHIST; ++t) {
    extract_inp_kernel<<<dim3((N_TRK * 32 + 255) / 256), dim3(256), 0, stream>>>(w.Tracks, ws.inp, t, 32);
    rm_lstm(stream, w, ws, ws.Sh);
    run_sp(stream, w, ws);
  }

  // ---- 10 prediction steps ----
  extract_inp_kernel<<<dim3((N_TRK * 4 + 255) / 256), dim3(256), 0, stream>>>(w.Tracks, ws.cf, LHIST - 1, 4);
  for (int t = 0; t < PLEN; ++t) {
    // x1 = relu(Sh @ W_p1.T + b_p1) + Sh
    gemm(stream, ws.Sh, HDIM, 0, w.W_p1, HDIM, 0, ws.x1, HDIM, 0,
         w.b_p1, ws.Sh, HDIM, N_TRK, HDIM, HDIM, 1.f, 1, 0, 1);
    // x2 = relu(x1 @ W_p2.T + b_p2) + x1
    gemm(stream, ws.x1, HDIM, 0, w.W_p2, HDIM, 0, ws.x2, HDIM, 0,
         w.b_p2, ws.x1, HDIM, N_TRK, HDIM, HDIM, 1.f, 1, 0, 1);
    // a = x2 @ W_pl.T + b_pl   (N x 2)
    gemm(stream, ws.x2, HDIM, 0, w.W_pl, HDIM, 0, ws.ab, 2, 0,
         w.b_pl, nullptr, 0, N_TRK, 2, HDIM, 1.f, 0, 0, 1);
    pstep_integrate_kernel<<<dim3((N_TRK + 255) / 256), dim3(256), 0, stream>>>(
        ws.ab, ws.cf, ws.inp, out, t);
    rm_lstm(stream, w, ws, ws.Sh);
    run_sp(stream, w, ws);
  }
}
